// CTRNN_38792144617814
// MI455X (gfx1250) — matrix operand order repaired and checked by reading, not compiled, not run on hardware
//
#include <hip/hip_runtime.h>

// CTRNN scan for MI455X (gfx1250, wave32).
// v5: weights fully pinned on-chip; A-fragment loads software-pipelined.
//   - bf16 B-fragments for k-steps 0..10 in registers (352 VGPRs/wave via
//     VGPR-MSB banks; WMMA reads high banks directly),
//   - k-steps 11..19 pinned in LDS (288 KB), 20 KB LDS A-exchange buffer,
//   - A fragments double-buffered one rank ahead (+8 VGPRs only — round 4
//     showed pipelining B too pushes past 512 VGPRs and spills breg to
//     scratch; the scheduler already pipelines the LDS-B loads on its own).
// Inner loop: 80 v_wmma_f32_16x16x32_bf16 per step, v_tanh_f32 epilogue,
// non-temporal 524 MB output stream, x prefetched 2 steps ahead.

typedef __attribute__((ext_vector_type(16))) __bf16 v16bf;
typedef __attribute__((ext_vector_type(8)))  float  v8f;

#define T_STEPS 1000
#define BATCH   256
#define IN_DIM  128
#define N_DIM   512
#define BT      16          // batch rows per block
#define KX      4           // 128/32 k-steps from the input projection
#define KSTEPS  20          // (128+512)/32 combined k-steps
#define KR      11          // k-steps whose B-fragments stay in registers
#define KL      (KSTEPS - KR) // 9 k-steps whose B-fragments stay in LDS
#define ALPHA   0.2f

__device__ __forceinline__ unsigned short f2bf(float f) {
  unsigned int u = __float_as_uint(f);
  u += 0x7FFFu + ((u >> 16) & 1u);      // round to nearest even
  return (unsigned short)(u >> 16);
}

__device__ __forceinline__ float fast_tanh(float xv) {
#if __has_builtin(__builtin_amdgcn_tanhf)
  return __builtin_amdgcn_tanhf(xv);    // native v_tanh_f32 (TRANS)
#else
  float ax = fabsf(xv);
  float t  = __expf(-2.0f * ax);
  float r  = (1.0f - t) / (1.0f + t);
  return xv < 0.0f ? -r : r;
#endif
}

// B-fragment layout (16x16x32 bf16): lane l holds column n = l%16;
// element e -> K = e (e<8) or e+8 (e>=8), plus 8 for lanes 16-31. B[k][n] = W[n][k].
// wfrag layout: [nt(32)][ks(20)][lane(32)][e(16)]  (bf16, 640 KB)
__global__ void __launch_bounds__(256)
pack_weights(const float* __restrict__ Win,
             const float* __restrict__ Whh,
             unsigned short* __restrict__ wfrag) {
  const int idx = blockIdx.x * 256 + threadIdx.x;
  if (idx >= 32 * KSTEPS * 32 * 16) return;
  const int e    = idx & 15;
  const int lane = (idx >> 4) & 31;
  const int rem  = idx >> 9;
  const int ks   = rem % KSTEPS;
  const int nt   = rem / KSTEPS;
  const int n    = nt * 16 + (lane & 15);
  const int kl   = e + ((e >> 3) << 3) + ((lane >> 4) << 3);
  float v;
  if (ks < KX) v = Win[n * IN_DIM + ks * 32 + kl];          // input proj part
  else         v = Whh[n * N_DIM + (ks - KX) * 32 + kl];    // recurrent part
  wfrag[idx] = f2bf(v);
}

// A-fragment layout (16x16x32 bf16): lane l holds row m = l%16;
// lanes 0-15:  e<8 -> K=e,    e>=8 -> K=e+8
// lanes 16-31: e<8 -> K=e+8,  e>=8 -> K=e+16
__global__ void __launch_bounds__(256, 1)
ctrnn_scan(const float* __restrict__ x,
           const float* __restrict__ h0,
           const float* __restrict__ b_in,
           const unsigned short* __restrict__ wfrag,
           float* __restrict__ out) {
  // LDS: 288 KB pinned B-fragments (ks 11..19) + 20 KB A exchange = 308 KB
  __shared__ __align__(32) unsigned short ldsB[KL * 32 * 32 * 16];
  __shared__ __align__(32) unsigned short ldsA[KSTEPS * 32 * 16];

  const int tid  = threadIdx.x;
  const int lane = tid & 31;
  const int wave = tid >> 5;
  const int hi   = lane >> 4;         // C/D row offset selector
  const int col  = lane & 15;         // C/D column within n-tile
  const int r0   = blockIdx.x * BT;   // first batch row owned by this block

  // ---- pin LDS-resident B-fragments: ldsB[ks'][nt][lane][e] <- wfrag ----
  {
    uint4*       dst = (uint4*)ldsB;
    const uint4* src = (const uint4*)wfrag;
#pragma unroll 8
    for (int i = 0; i < 72; ++i) {            // 18432 uint4 / 256 threads
      const int f   = i * 256 + tid;
      const int fd  = f >> 6;                 // fragment id (64 uint4 each)
      const int o   = f & 63;
      const int ks2 = fd >> 5;                // 0..KL-1
      const int nt  = fd & 31;
      dst[f] = src[((nt * KSTEPS + (ks2 + KR)) << 6) + o];
    }
  }

  // ---- register-resident B-fragments: 4 ntiles x KR ksteps ----
  v16bf breg[4][KR];
#pragma unroll
  for (int j = 0; j < 4; ++j) {
    const int ntj = wave * 4 + j;
#pragma unroll
    for (int k = 0; k < KR; ++k)
      breg[j][k] = *(const v16bf*)&wfrag[((ntj * KSTEPS + k) * 32 + lane) * 16];
  }

  float bias[4];
  v8f   hst[4];                       // f32 master copy of h (D layout)
#pragma unroll
  for (int j = 0; j < 4; ++j) {
    const int n = wave * 64 + j * 16 + col;
    bias[j] = b_in[n];
#pragma unroll
    for (int v = 0; v < 8; ++v)
      hst[j][v] = h0[(r0 + v + 8 * hi) * N_DIM + n];
  }

  // stage h0 as bf16 A-fragments (K region 128..639)
#pragma unroll
  for (int j = 0; j < 4; ++j) {
    const int n  = wave * 64 + j * 16 + col;   // K index in h part
    const int ks = KX + (n >> 5);
    const int r  = n & 31;
    const int lh = (r >> 3) & 1;
    const int e  = (r & 7) | ((r >> 4) << 3);
#pragma unroll
    for (int v = 0; v < 8; ++v) {
      const int m = v + 8 * hi;
      ldsA[((ks * 32) + (m + 16 * lh)) * 16 + e] = f2bf(hst[j][v]);
    }
  }

  // stage x(t=0) as bf16 A-fragments (K region 0..127)
#pragma unroll
  for (int p = 0; p < 8; ++p) {
    const int f  = p * 256 + tid;
    const int e  = f & 15;
    const int la = (f >> 4) & 31;
    const int ks = f >> 9;
    const int m  = la & 15;
    const int kl = e + ((e >> 3) << 3) + ((la >> 4) << 3);
    ldsA[((ks * 32) + la) * 16 + e] = f2bf(x[(r0 + m) * IN_DIM + ks * 32 + kl]);
  }
  __syncthreads();

  for (int t = 0; t < T_STEPS; ++t) {
    v8f acc[4] = {};   // zero accumulators; bias folded in at the epilogue

    // prefetch x tile for step t+2 into caches (one line-touch per thread)
    if (t + 2 < T_STEPS) {
      const char* p =
          (const char*)&x[(size_t)((t + 2) * BATCH + r0) * IN_DIM] + tid * 32;
      __builtin_prefetch(p, 0, 1);
    }

    // fused (x @ W_in^T + h @ W_hh^T): 20 ranks x 4 ntiles of bf16 WMMA.
    // A fragments double-buffered one rank ahead; B from registers
    // (ks < KR) or pinned LDS (scheduler pipelines those loads itself).
    v16bf a_cur = *(const v16bf*)&ldsA[lane * 16];
#pragma unroll
    for (int ks = 0; ks < KSTEPS; ++ks) {
      v16bf a_nxt;
      if (ks + 1 < KSTEPS)
        a_nxt = *(const v16bf*)&ldsA[((ks + 1) * 32 + lane) * 16];
#pragma unroll
      for (int j = 0; j < 4; ++j) {
        v16bf b;
        if (ks < KR) {
          b = breg[j][ks];
        } else {
          b = *(const v16bf*)
              &ldsB[(((ks - KR) * 32) + (wave * 4 + j)) * 512 + lane * 16];
        }
        acc[j] = __builtin_amdgcn_wmma_f32_16x16x32_bf16(
            false, a_cur, false, b, (short)0, acc[j], false, false);
      }
      if (ks + 1 < KSTEPS) a_cur = a_nxt;
    }

    // h = (1-a)*h + a*tanh(acc + bias); stream to out[t] (non-temporal)
#pragma unroll
    for (int j = 0; j < 4; ++j) {
      const int n = wave * 64 + j * 16 + col;
#pragma unroll
      for (int v = 0; v < 8; ++v) {
        const float hn = fast_tanh(acc[j][v] + bias[j]);
        const float h  = hst[j][v] * (1.0f - ALPHA) + ALPHA * hn;
        hst[j][v] = h;
        __builtin_nontemporal_store(
            h, &out[(size_t)(t * BATCH + r0 + v + 8 * hi) * N_DIM + n]);
      }
    }

    __syncthreads();   // all waves done reading ldsA for step t

    // stage h(t) fragments for step t+1 (each wave owns its 64 columns)
#pragma unroll
    for (int j = 0; j < 4; ++j) {
      const int n  = wave * 64 + j * 16 + col;
      const int ks = KX + (n >> 5);
      const int r  = n & 31;
      const int lh = (r >> 3) & 1;
      const int e  = (r & 7) | ((r >> 4) << 3);
#pragma unroll
      for (int v = 0; v < 8; ++v) {
        const int m = v + 8 * hi;
        ldsA[((ks * 32) + (m + 16 * lh)) * 16 + e] = f2bf(hst[j][v]);
      }
    }

    if (t + 1 < T_STEPS) {   // stage x(t+1)
#pragma unroll
      for (int p = 0; p < 8; ++p) {
        const int f  = p * 256 + tid;
        const int e  = f & 15;
        const int la = (f >> 4) & 31;
        const int ks = f >> 9;
        const int m  = la & 15;
        const int kl = e + ((e >> 3) << 3) + ((la >> 4) << 3);
        ldsA[((ks * 32) + la) * 16 + e] =
            f2bf(x[(size_t)((t + 1) * BATCH + r0 + m) * IN_DIM + ks * 32 + kl]);
      }
    }
    __syncthreads();
  }

  // final hidden state -> tail of d_out
#pragma unroll
  for (int j = 0; j < 4; ++j) {
    const int n = wave * 64 + j * 16 + col;
#pragma unroll
    for (int v = 0; v < 8; ++v)
      out[(size_t)T_STEPS * BATCH * N_DIM + (r0 + v + 8 * hi) * N_DIM + n] =
          hst[j][v];
  }
}

extern "C" void kernel_launch(void* const* d_in, const int* in_sizes, int n_in,
                              void* d_out, int out_size, void* d_ws,
                              size_t ws_size, hipStream_t stream) {
  const float* x   = (const float*)d_in[0];   // [1000,256,128]
  const float* h0  = (const float*)d_in[1];   // [256,512]
  const float* Win = (const float*)d_in[2];   // [512,128]
  const float* bin = (const float*)d_in[3];   // [512]
  const float* Whh = (const float*)d_in[4];   // [512,512]
  unsigned short* wfrag = (unsigned short*)d_ws;  // needs 640 KB

  const int packElems = 32 * KSTEPS * 32 * 16;    // 327680
  pack_weights<<<(packElems + 255) / 256, 256, 0, stream>>>(Win, Whh, wfrag);
  ctrnn_scan<<<BATCH / BT, 256, 0, stream>>>(x, h0, bin, wfrag, (float*)d_out);
}
